// LinearCrossAttention_86500641342072
// MI455X (gfx1250) — matrix operand order repaired and checked by reading, not compiled, not run on hardware
//
#include <hip/hip_runtime.h>
#include <hip/hip_bf16.h>

typedef __attribute__((ext_vector_type(16))) _Float16 v16h;
typedef __attribute__((ext_vector_type(8)))  _Float16 v8h;
typedef __attribute__((ext_vector_type(8)))  float    v8f;

#define HEADS     8
#define DIM_HEAD  64
#define HIDDEN    512
#define DIM_      512
#define LABEL_DIM 128
#define NPIX      4096
#define BATCH     16
#define EPS_      1e-5f

// ---------------- WMMA fragment helpers (16x16x32 f16, wave32) ----------------
// A-tile is stored in LDS with K permuted inside each 32-chunk: groups of 8
// reordered {0,2,1,3}, so each lane's 16 fragment halves are contiguous+aligned.
__device__ inline int kperm(int c) {           // c in [0,32)
  int g  = (c >> 3) & 3;
  int ng = ((g & 1) << 1) | (g >> 1);          // 0->0, 1->2, 2->1, 3->3
  return (c & 7) | (ng << 3);
}
// A fragment 16x32 (MxK): lanes 0-15 -> M=lane, K {0..7,16..23};
//                         lanes 16-31 -> M=lane-16, K {8..15,24..31}
__device__ inline v16h frag_a(const _Float16* tile, int ld, int row0, int lane) {
  const _Float16* p = tile + (row0 + (lane & 15)) * ld + ((lane >> 4) << 4);
  union { v16h v; v8h h[2]; } u;
  u.h[0] = *(const v8h*)p;
  u.h[1] = *(const v8h*)(p + 8);
  return u.v;
}
// B fragment 32x16 (KxN): lanes 0-15 -> N=lane, K {0..7,16..23};
//                         lanes 16-31 -> N=lane-16, K {8..15,24..31}
__device__ inline v16h frag_b(const _Float16* tile, int ld, int col0, int lane) {
  const _Float16* p = tile + col0 + (lane & 15) + ((lane >> 4) << 3) * ld;
  v16h v;
#pragma unroll
  for (int s = 0; s < 16; ++s) v[s] = p[((s & 7) + ((s >> 3) << 4)) * ld];
  return v;
}

// ---------------- Kernel 1: QKV = Wqkv[1536x512] @ X[b][512x4096], f16 out ----
__global__ void __launch_bounds__(256) k_qkv_gemm(const float* __restrict__ X,
                                                  const float* __restrict__ W,
                                                  _Float16* __restrict__ QKV) {
  const int M = 3 * HIDDEN, K = DIM_, N = NPIX;
  __shared__ _Float16 sA[64 * 32];
  __shared__ _Float16 sB[32 * 128];
  const int b  = blockIdx.z;
  const int m0 = blockIdx.y * 64;
  const int n0 = blockIdx.x * 128;
  const int tid = threadIdx.x, lane = tid & 31, wid = tid >> 5;
  const int wm = (wid >> 2) * 32, wn = (wid & 3) * 32;
  // base pointers; advanced by constants each K-step -> immediate-offset loads
  const float* pa = W + (size_t)(m0 + (tid >> 5)) * K + (tid & 31);
  const float* pb = X + (size_t)b * K * N + (size_t)(tid >> 7) * N + n0 + (tid & 127);
  _Float16* sa_st = &sA[(tid >> 5) * 32 + kperm(tid & 31)];
  _Float16* sb_st = &sB[tid];
  v8f acc[2][2] = {};
  for (int k0 = 0; k0 < K; k0 += 32) {
#pragma unroll
    for (int i = 0; i < 8; ++i)  sa_st[i * 256] = (_Float16)pa[(size_t)i * 8 * K];
#pragma unroll
    for (int i = 0; i < 16; ++i) sb_st[i * 256] = (_Float16)pb[(size_t)i * 2 * N];
    if (k0 + 32 < K) {                           // gfx1250 global_prefetch
      __builtin_prefetch(pa + 32, 0, 1);
      __builtin_prefetch(pb + (size_t)32 * N, 0, 1);
    }
    pa += 32;
    pb += (size_t)32 * N;
    __syncthreads();
#pragma unroll
    for (int mi = 0; mi < 2; ++mi) {
      v16h a = frag_a(sA, 32, wm + mi * 16, lane);
#pragma unroll
      for (int ni = 0; ni < 2; ++ni) {
        v16h bf = frag_b(sB, 128, wn + ni * 16, lane);
        acc[mi][ni] = __builtin_amdgcn_wmma_f32_16x16x32_f16(false, a, false, bf,
                                                             (short)0, acc[mi][ni], false, false);
      }
    }
    __syncthreads();
  }
  _Float16* Cb = QKV + (size_t)b * M * N;
  const int col  = n0 + wn + (lane & 15);
  const int rtop = m0 + wm + ((lane >> 4) << 3);
#pragma unroll
  for (int mi = 0; mi < 2; ++mi)
#pragma unroll
    for (int ni = 0; ni < 2; ++ni) {
      _Float16* cp = Cb + (size_t)(rtop + mi * 16) * N + col + ni * 16;
#pragma unroll
      for (int r = 0; r < 8; ++r) cp[(size_t)r * N] = (_Float16)acc[mi][ni][r];
    }
}

// ---------------- Kernel 2: label token projections lk/lv [16x512] ------------
__global__ void __launch_bounds__(256) k_label_proj(const float* __restrict__ lab,
                                                    const float* __restrict__ wlk,
                                                    const float* __restrict__ blk,
                                                    const float* __restrict__ wlv,
                                                    const float* __restrict__ blv,
                                                    float* __restrict__ lk,
                                                    float* __restrict__ lv) {
  int idx = blockIdx.x * 256 + threadIdx.x;
  if (idx >= BATCH * HIDDEN) return;
  int b = idx >> 9, o = idx & 511;
  const float* l = lab + b * LABEL_DIM;
  float sk = blk[o], sv = blv[o];
  for (int c = 0; c < LABEL_DIM; ++c) {
    float lc = l[c];
    sk += lc * wlk[o * LABEL_DIM + c];
    sv += lc * wlv[o * LABEL_DIM + c];
  }
  lk[idx] = sk;
  lv[idx] = sv;
}

// ---------------- Kernel 3: softmax over d (64) on Q, *scale, in place --------
__global__ void __launch_bounds__(256) k_q_softmax(_Float16* __restrict__ QKV) {
  size_t idx = (size_t)blockIdx.x * 256 + threadIdx.x;  // BATCH*HEADS*NPIX
  int n = (int)(idx & (NPIX - 1));
  int t = (int)(idx >> 12);          // b*HEADS + h
  int h = t & (HEADS - 1);
  int b = t >> 3;
  _Float16* base = QKV + ((size_t)(b * 3 * HIDDEN + h * DIM_HEAD)) * NPIX + n;
  float mx = -1e30f;
  for (int d = 0; d < DIM_HEAD; ++d)
    mx = fmaxf(mx, (float)base[(size_t)d * NPIX]);
  float s = 0.f;
  for (int d = 0; d < DIM_HEAD; ++d)
    s += __expf((float)base[(size_t)d * NPIX] - mx);
  float inv = 0.125f / s;            // scale = DIM_HEAD^-0.5 = 0.125 folded in
  for (int d = 0; d < DIM_HEAD; ++d) {
    float v = __expf((float)base[(size_t)d * NPIX] - mx) * inv;
    base[(size_t)d * NPIX] = (_Float16)v;
  }
}

// ---------------- Kernel 4: per-row max / 1/sum(exp) over n (incl. label) -----
__global__ void __launch_bounds__(256) k_k_stats(const _Float16* __restrict__ QKV,
                                                 const float* __restrict__ lk,
                                                 float* __restrict__ kmax,
                                                 float* __restrict__ kinv) {
  __shared__ float red[256];
  int row = blockIdx.x;              // 0..8191 = b*512 + o
  int b = row >> 9, o = row & 511;
  const _Float16* base = QKV + ((size_t)(b * 3 * HIDDEN + HIDDEN + o)) * NPIX;
  int tid = threadIdx.x;
  float vals[16];
  float mx = -1e30f;
#pragma unroll
  for (int i = 0; i < 16; ++i) {
    vals[i] = (float)base[tid + i * 256];
    mx = fmaxf(mx, vals[i]);
  }
  red[tid] = mx; __syncthreads();
  for (int s2 = 128; s2 > 0; s2 >>= 1) {
    if (tid < s2) red[tid] = fmaxf(red[tid], red[tid + s2]);
    __syncthreads();
  }
  float lkv = lk[b * HIDDEN + o];
  mx = fmaxf(red[0], lkv);
  __syncthreads();
  float s = 0.f;
#pragma unroll
  for (int i = 0; i < 16; ++i) s += __expf(vals[i] - mx);
  red[tid] = s; __syncthreads();
  for (int s2 = 128; s2 > 0; s2 >>= 1) {
    if (tid < s2) red[tid] += red[tid + s2];
    __syncthreads();
  }
  if (tid == 0) {
    float tot = red[0] + __expf(lkv - mx);
    kmax[row] = mx;
    kinv[row] = 1.0f / tot;
  }
}

// ---------------- Kernel 5: context[b,h,d,e] = sum_n knorm(d,n) v(e,n) --------
__global__ void __launch_bounds__(256) k_context(const _Float16* __restrict__ QKV,
                                                 const float* __restrict__ kmax,
                                                 const float* __restrict__ kinv,
                                                 const float* __restrict__ lk,
                                                 const float* __restrict__ lv,
                                                 float* __restrict__ CTX) {
  __shared__ _Float16 sA[64 * 32];   // normalized K: [d][kperm(kk)]
  __shared__ _Float16 sB[32 * 64];   // V^T: [kk][e]
  int bh = blockIdx.x;
  int b = bh >> 3, h = bh & 7;
  int tid = threadIdx.x, lane = tid & 31, wid = tid >> 5;
  int wm = (wid >> 1) * 16, wn = (wid & 1) * 32;
  const _Float16* Kb = QKV + ((size_t)(b * 3 * HIDDEN +     HIDDEN + h * DIM_HEAD)) * NPIX;
  const _Float16* Vb = QKV + ((size_t)(b * 3 * HIDDEN + 2 * HIDDEN + h * DIM_HEAD)) * NPIX;
  const float* km = kmax + b * HIDDEN + h * DIM_HEAD;
  const float* ki = kinv + b * HIDDEN + h * DIM_HEAD;
  // per-thread bases, advanced by 32 columns per step
  const int drow = tid >> 5, kk = tid & 31;
  const _Float16* pk = Kb + (size_t)drow * NPIX + kk;
  const _Float16* pv = Vb + (size_t)drow * NPIX + kk;
  _Float16* sa_st = &sA[drow * 32 + kperm(kk)];
  _Float16* sb_st = &sB[kk * 64 + drow];
  float kmx[8], kin[8];
#pragma unroll
  for (int i = 0; i < 8; ++i) { kmx[i] = km[drow + i * 8]; kin[i] = ki[drow + i * 8]; }
  v8f acc[2] = {};
  for (int n0 = 0; n0 < NPIX; n0 += 32) {
#pragma unroll
    for (int i = 0; i < 8; ++i) {                 // A: 64(d) x 32(kk), normalized
      float raw = (float)pk[(size_t)i * 8 * NPIX];
      sa_st[i * 256] = (_Float16)(__expf(raw - kmx[i]) * kin[i]);
    }
#pragma unroll
    for (int i = 0; i < 8; ++i)                   // B: V transposed to [kk][e]
      sb_st[i * 8] = pv[(size_t)i * 8 * NPIX];
    pk += 32;
    pv += 32;
    __syncthreads();
    v16h a = frag_a(sA, 32, wm, lane);
#pragma unroll
    for (int ni = 0; ni < 2; ++ni) {
      v16h bf = frag_b(sB, 64, wn + ni * 16, lane);
      acc[ni] = __builtin_amdgcn_wmma_f32_16x16x32_f16(false, a, false, bf,
                                                       (short)0, acc[ni], false, false);
    }
    __syncthreads();
  }
  float* C = CTX + (size_t)bh * 64 * 64;          // [d][e]
#pragma unroll
  for (int ni = 0; ni < 2; ++ni) {
    int e = wn + ni * 16 + (lane & 15);
    float lvv = lv[b * HIDDEN + h * DIM_HEAD + e];
    int dbase = wm + ((lane >> 4) << 3);
#pragma unroll
    for (int r = 0; r < 8; ++r) {
      int d = dbase + r;
      float kl = __expf(lk[b * HIDDEN + h * DIM_HEAD + d] - km[d]) * ki[d];
      C[d * 64 + e] = acc[ni][r] + kl * lvv;      // label token contribution
    }
  }
}

// ---------------- Kernel 6: attn[e,n] = sum_d ctx[d,e] * q[d,n] ---------------
__global__ void __launch_bounds__(256) k_attn_out(const float* __restrict__ CTX,
                                                  const _Float16* __restrict__ QKV,
                                                  _Float16* __restrict__ ATT) {
  __shared__ _Float16 sA[64 * 64];   // ctx^T: [e][kperm within 32-chunks of d]
  __shared__ _Float16 sB[32 * 128];
  int bh = blockIdx.y;
  int b = bh >> 3, h = bh & 7;
  int n0 = blockIdx.x * 128;
  int tid = threadIdx.x, lane = tid & 31, wid = tid >> 5;
  int wm = (wid >> 2) * 32, wn = (wid & 3) * 32;
  const float* C = CTX + (size_t)bh * 64 * 64;
#pragma unroll
  for (int i = 0; i < 16; ++i) {                  // transpose ctx into LDS once
    int idx = tid + i * 256;
    int d = idx >> 6, e = idx & 63;
    sA[e * 64 + (d & 32) + kperm(d & 31)] = (_Float16)C[idx];
  }
  const _Float16* Qb = QKV + ((size_t)(b * 3 * HIDDEN + h * DIM_HEAD)) * NPIX;
  const _Float16* pq = Qb + (size_t)(tid >> 7) * NPIX + n0 + (tid & 127);
  _Float16* sb_st = &sB[tid];
  v8f acc[2][2] = {};
  for (int k0 = 0; k0 < 64; k0 += 32) {
#pragma unroll
    for (int i = 0; i < 16; ++i)                  // B tile 32x128 from softmaxed q
      sb_st[i * 256] = pq[(size_t)i * 2 * NPIX];
    pq += (size_t)32 * NPIX;
    __syncthreads();
#pragma unroll
    for (int mi = 0; mi < 2; ++mi) {
      v16h a = frag_a(sA + k0, 64, wm + mi * 16, lane);
#pragma unroll
      for (int ni = 0; ni < 2; ++ni) {
        v16h bf = frag_b(sB, 128, wn + ni * 16, lane);
        acc[mi][ni] = __builtin_amdgcn_wmma_f32_16x16x32_f16(false, a, false, bf,
                                                             (short)0, acc[mi][ni], false, false);
      }
    }
    __syncthreads();
  }
  _Float16* Ob = ATT + ((size_t)(b * HIDDEN + h * DIM_HEAD)) * NPIX;
  const int col  = n0 + wn + (lane & 15);
  const int rtop = wm + ((lane >> 4) << 3);
#pragma unroll
  for (int mi = 0; mi < 2; ++mi)
#pragma unroll
    for (int ni = 0; ni < 2; ++ni) {
      _Float16* cp = Ob + (size_t)(rtop + mi * 16) * NPIX + col + ni * 16;
#pragma unroll
      for (int r = 0; r < 8; ++r) cp[(size_t)r * NPIX] = (_Float16)acc[mi][ni][r];
    }
}

// ---------------- Kernel 7: out2 = Wout @ attn + bias, + GN partial sums ------
__global__ void __launch_bounds__(256) k_out_gemm(const _Float16* __restrict__ ATT,
                                                  const float* __restrict__ W,
                                                  const float* __restrict__ bias,
                                                  float* __restrict__ OUT2,
                                                  float* __restrict__ part) {
  const int M = DIM_, K = HIDDEN, N = NPIX;
  __shared__ _Float16 sA[64 * 32];
  __shared__ _Float16 sB[32 * 128];
  __shared__ float red[256];
  int b  = blockIdx.z;
  int m0 = blockIdx.y * 64;
  int n0 = blockIdx.x * 128;
  int tid = threadIdx.x, lane = tid & 31, wid = tid >> 5;
  int wm = (wid >> 2) * 32, wn = (wid & 3) * 32;
  const float*    pa = W + (size_t)(m0 + (tid >> 5)) * K + (tid & 31);
  const _Float16* pb = ATT + (size_t)b * K * N + (size_t)(tid >> 7) * N + n0 + (tid & 127);
  _Float16* sa_st = &sA[(tid >> 5) * 32 + kperm(tid & 31)];
  _Float16* sb_st = &sB[tid];
  v8f acc[2][2] = {};
  for (int k0 = 0; k0 < K; k0 += 32) {
#pragma unroll
    for (int i = 0; i < 8; ++i)  sa_st[i * 256] = (_Float16)pa[(size_t)i * 8 * K];
#pragma unroll
    for (int i = 0; i < 16; ++i) sb_st[i * 256] = pb[(size_t)i * 2 * N];
    if (k0 + 32 < K) {
      __builtin_prefetch(pa + 32, 0, 1);
      __builtin_prefetch(pb + (size_t)32 * N, 0, 1);
    }
    pa += 32;
    pb += (size_t)32 * N;
    __syncthreads();
#pragma unroll
    for (int mi = 0; mi < 2; ++mi) {
      v16h a = frag_a(sA, 32, wm + mi * 16, lane);
#pragma unroll
      for (int ni = 0; ni < 2; ++ni) {
        v16h bf = frag_b(sB, 128, wn + ni * 16, lane);
        acc[mi][ni] = __builtin_amdgcn_wmma_f32_16x16x32_f16(false, a, false, bf,
                                                             (short)0, acc[mi][ni], false, false);
      }
    }
    __syncthreads();
  }
  float lsum = 0.f, lsq = 0.f;
  float* Ob = OUT2 + (size_t)b * M * N;
  const int col  = n0 + wn + (lane & 15);
  const int rtop = m0 + wm + ((lane >> 4) << 3);
#pragma unroll
  for (int mi = 0; mi < 2; ++mi)
#pragma unroll
    for (int ni = 0; ni < 2; ++ni) {
      float* cp = Ob + (size_t)(rtop + mi * 16) * N + col + ni * 16;
#pragma unroll
      for (int r = 0; r < 8; ++r) {
        float v = acc[mi][ni][r] + bias[rtop + mi * 16 + r];
        cp[(size_t)r * N] = v;
        lsum += v; lsq += v * v;
      }
    }
  red[tid] = lsum; __syncthreads();
  for (int s2 = 128; s2 > 0; s2 >>= 1) { if (tid < s2) red[tid] += red[tid + s2]; __syncthreads(); }
  float bsum = red[0]; __syncthreads();
  red[tid] = lsq; __syncthreads();
  for (int s2 = 128; s2 > 0; s2 >>= 1) { if (tid < s2) red[tid] += red[tid + s2]; __syncthreads(); }
  if (tid == 0) {
    int pi = ((b * 8 + blockIdx.y) * 32 + blockIdx.x) * 2;
    part[pi]     = bsum;
    part[pi + 1] = red[0];
  }
}

// ---------------- Kernel 8: reduce partials -> mean / rstd per batch ----------
__global__ void __launch_bounds__(256) k_gn_stats(const float* __restrict__ part,
                                                  float* __restrict__ stats) {
  __shared__ float rs[256], rq[256];
  int b = blockIdx.x, tid = threadIdx.x;
  const float* p = part + b * 512;                // 256 (sum,sumsq) pairs
  rs[tid] = p[tid * 2];
  rq[tid] = p[tid * 2 + 1];
  __syncthreads();
  for (int s2 = 128; s2 > 0; s2 >>= 1) {
    if (tid < s2) { rs[tid] += rs[tid + s2]; rq[tid] += rq[tid + s2]; }
    __syncthreads();
  }
  if (tid == 0) {
    const float cnt = (float)DIM_ * (float)NPIX;
    float mean = rs[0] / cnt;
    float var  = rq[0] / cnt - mean * mean;
    stats[b * 2]     = mean;
    stats[b * 2 + 1] = rsqrtf(var + EPS_);
  }
}

// ---------------- Kernel 9: apply GroupNorm affine ----------------------------
__global__ void __launch_bounds__(256) k_gn_apply(const float* __restrict__ OUT2,
                                                  const float* __restrict__ stats,
                                                  const float* __restrict__ g,
                                                  const float* __restrict__ beta,
                                                  float* __restrict__ out) {
  size_t idx = (size_t)blockIdx.x * 256 + threadIdx.x;  // 16*512*4096
  int b = (int)(idx >> 21);
  int c = (int)((idx >> 12) & 511);
  float mean = stats[b * 2], rstd = stats[b * 2 + 1];
  out[idx] = (OUT2[idx] - mean) * rstd * g[c] + beta[c];
}

// ---------------- host-side launch --------------------------------------------
extern "C" void kernel_launch(void* const* d_in, const int* in_sizes, int n_in,
                              void* d_out, int out_size, void* d_ws, size_t ws_size,
                              hipStream_t stream) {
  (void)in_sizes; (void)n_in; (void)out_size; (void)ws_size;
  const float* x    = (const float*)d_in[0];
  const float* lab  = (const float*)d_in[1];
  const float* wqkv = (const float*)d_in[2];
  const float* wlk  = (const float*)d_in[3];
  const float* blk  = (const float*)d_in[4];
  const float* wlv  = (const float*)d_in[5];
  const float* blv  = (const float*)d_in[6];
  const float* wout = (const float*)d_in[7];
  const float* bout = (const float*)d_in[8];
  const float* gng  = (const float*)d_in[9];
  const float* gnb  = (const float*)d_in[10];
  float* out = (float*)d_out;

  char* ws = (char*)d_ws;
  size_t off = 0;
  _Float16* qkv = (_Float16*)(ws + off); off += (size_t)BATCH * 3 * HIDDEN * NPIX * 2; // 201 MB
  _Float16* att = (_Float16*)(ws + off); off += (size_t)BATCH * HIDDEN * NPIX * 2;     // 67 MB
  float* out2   = (float*)(ws + off);    off += (size_t)BATCH * DIM_ * NPIX * 4;       // 134 MB
  float* ctx    = (float*)(ws + off);    off += (size_t)BATCH * HEADS * 64 * 64 * 4;   // 2 MB
  float* lk     = (float*)(ws + off);    off += (size_t)BATCH * HIDDEN * 4;
  float* lv     = (float*)(ws + off);    off += (size_t)BATCH * HIDDEN * 4;
  float* kmax   = (float*)(ws + off);    off += (size_t)BATCH * HIDDEN * 4;
  float* kinv   = (float*)(ws + off);    off += (size_t)BATCH * HIDDEN * 4;
  float* part   = (float*)(ws + off);    off += (size_t)BATCH * 256 * 2 * 4;
  float* stats  = (float*)(ws + off);    off += (size_t)BATCH * 2 * 4;

  k_qkv_gemm  <<<dim3(NPIX / 128, (3 * HIDDEN) / 64, BATCH), 256, 0, stream>>>(x, wqkv, qkv);
  k_label_proj<<<dim3((BATCH * HIDDEN + 255) / 256), 256, 0, stream>>>(lab, wlk, blk, wlv, blv, lk, lv);
  k_q_softmax <<<dim3((BATCH * HEADS * NPIX) / 256), 256, 0, stream>>>(qkv);
  k_k_stats   <<<dim3(BATCH * HIDDEN), 256, 0, stream>>>(qkv, lk, kmax, kinv);
  k_context   <<<dim3(BATCH * HEADS), 256, 0, stream>>>(qkv, kmax, kinv, lk, lv, ctx);
  k_attn_out  <<<dim3(NPIX / 128, BATCH * HEADS), 256, 0, stream>>>(ctx, qkv, att);
  k_out_gemm  <<<dim3(NPIX / 128, DIM_ / 64, BATCH), 256, 0, stream>>>(att, wout, bout, out2, part);
  k_gn_stats  <<<dim3(BATCH), 256, 0, stream>>>(part, stats);
  k_gn_apply  <<<dim3((size_t)BATCH * DIM_ * NPIX / 256), 256, 0, stream>>>(out2, stats, gng, gnb, out);
}
